// SRNN_11725260718425
// MI455X (gfx1250) — matrix operand order repaired and checked by reading, not compiled, not run on hardware
//
#include <hip/hip_runtime.h>

// ---------------- CDNA5 WMMA fragment types ----------------
typedef __attribute__((ext_vector_type(16))) _Float16 v16h;
typedef __attribute__((ext_vector_type(8)))  float    v8f;

#define LL 512
#define BB 512
#define SS 64

// ---- swizzled f16 weight arena offsets (in halves), lives in d_ws ----
enum : int {
  OFF_PU_W1  = 0,        // 4x1 tiles
  OFF_PU_W2  = 2048,     // 4x2
  OFF_PX_W1  = 6144,     // 4x1
  OFF_PX_W2  = 8192,     // 4x2
  OFF_WIH    = 12288,    // 4x2
  OFF_WHH    = 16384,    // 4x2
  OFF_A_W1   = 20480,    // 4x4
  OFF_A_W2   = 28672,    // 4x2
  OFF_PO_FC1 = 32768,    // 8x4
  OFF_PO_FC2 = 49152,    // 8x4
  OFF_PO_MU  = 65536,    // 4x4
  OFF_PO_LS  = 73728,    // 4x4
  OFF_PR_FC1 = 81920,    // 8x4
  OFF_PR_FC2 = 98304,    // 8x4
  OFF_PR_MU  = 114688,   // 4x4
  SW_TOTAL   = 122880    // halves (240 KB)
};
#define HALF8  (4 * 4 * 512)   // tile offset of n-tiles 4..7 in an 8x4 matrix

struct Params {
  const float* __restrict__ ext;
  const float* __restrict__ obs;
  const float* __restrict__ noise;
  const float* __restrict__ pu_b1;  const float* __restrict__ pu_b2;
  const float* __restrict__ px_b1;  const float* __restrict__ px_b2;
  const float* __restrict__ rnn_bih; const float* __restrict__ rnn_bhh;
  const float* __restrict__ a_b1;   const float* __restrict__ a_b2;
  const float* __restrict__ po_fc1_b; const float* __restrict__ po_fc2_b;
  const float* __restrict__ po_mu_b;  const float* __restrict__ po_ls_b;
  const float* __restrict__ pr_fc1_b; const float* __restrict__ pr_fc2_b;
  const float* __restrict__ pr_mu_b;
  const _Float16* __restrict__ ws;   // swizzled f16 weights
  float* __restrict__ out_mu; float* __restrict__ out_ls;
  float* __restrict__ out_z;  float* __restrict__ out_d;
};

__device__ inline int laneid() { return threadIdx.x & 31; }

// ---- fast transcendentals: CDNA5 has V_TANH_F32 / V_RCP_F32 / V_EXP_F32 ----
__device__ inline float fast_tanh(float x) {
#if __has_builtin(__builtin_amdgcn_tanhf)
  return __builtin_amdgcn_tanhf(x);
#elif __has_builtin(__builtin_amdgcn_tanh_f32)
  return __builtin_amdgcn_tanh_f32(x);
#else
  return tanhf(x);
#endif
}
__device__ inline float fast_sigmoid(float x) {
  return __builtin_amdgcn_rcpf(1.0f + __expf(-x));
}

// ---------- fragment-order loads: one 32B vector load per fragment ----------
__device__ inline v16h ldA(const _Float16* ch) {
  return *(const v16h*)(ch + laneid() * 16);
}
__device__ inline v16h ldB(const _Float16* __restrict__ w) {
  return *(const v16h*)(w + laneid() * 16);
}

// A-fragment gather from a global f32 [16 x 16] tile (preprocess fc1, K=16->32 pad).
__device__ inline v16h ldA_g16(const float* __restrict__ s) {
  const int l = laneid(), m = l & 15, hi = l >> 4;
  v16h a;
#pragma unroll
  for (int j = 0; j < 16; ++j) {
    const int v = j >> 1, p = j & 1;
    const int k = ((v & 4) ? 16 : 0) + (hi ? 8 : 0) + 2 * (v & 3) + p;
    a[j] = (_Float16)((k < 16) ? s[m * 16 + k] : 0.0f);
  }
  return a;
}

// ---------- A source = up to 4 chunks (handles 64|64 concats) ----------
struct ASrc { const _Float16* ch[4]; };
__device__ inline ASrc src64(const _Float16* b) {
  ASrc s; s.ch[0] = b; s.ch[1] = b + 512; s.ch[2] = b; s.ch[3] = b; return s;
}
__device__ inline ASrc cat64(const _Float16* b0, const _Float16* b1) {
  ASrc s; s.ch[0] = b0; s.ch[1] = b0 + 512; s.ch[2] = b1; s.ch[3] = b1 + 512; return s;
}
__device__ inline ASrc src128(const _Float16* b) {
  ASrc s; s.ch[0] = b; s.ch[1] = b + 512; s.ch[2] = b + 1024; s.ch[3] = b + 1536; return s;
}

// ---------- accumulator init from pre-loaded per-lane bias registers ----------
template<int NT>
__device__ inline void acc_init_r(v8f* acc, const float* b) {
#pragma unroll
  for (int nt = 0; nt < NT; ++nt)
#pragma unroll
    for (int r = 0; r < 8; ++r) acc[nt][r] = b[nt];
}

// ---------- WMMA K-loops ----------
template<int NT, int KT>
__device__ inline void mma_loop(v8f* acc, const ASrc& a, const _Float16* __restrict__ w) {
#pragma unroll
  for (int kc = 0; kc < KT; ++kc) {
    const v16h A = ldA(a.ch[kc]);
#pragma unroll
    for (int nt = 0; nt < NT; ++nt) {
      const v16h B = ldB(w + (size_t)(nt * KT + kc) * 512);
      acc[nt] = __builtin_amdgcn_wmma_f32_16x16x32_f16(
          false, A, false, B, (short)0, acc[nt], false, false);
    }
  }
}
// two GEMMs sharing A fragments
template<int NT, int KT>
__device__ inline void mma_loop2(v8f* accA, v8f* accB, const ASrc& a,
                                 const _Float16* __restrict__ wA,
                                 const _Float16* __restrict__ wB) {
#pragma unroll
  for (int kc = 0; kc < KT; ++kc) {
    const v16h A = ldA(a.ch[kc]);
#pragma unroll
    for (int nt = 0; nt < NT; ++nt) {
      const v16h B0 = ldB(wA + (size_t)(nt * KT + kc) * 512);
      accA[nt] = __builtin_amdgcn_wmma_f32_16x16x32_f16(
          false, A, false, B0, (short)0, accA[nt], false, false);
      const v16h B1 = ldB(wB + (size_t)(nt * KT + kc) * 512);
      accB[nt] = __builtin_amdgcn_wmma_f32_16x16x32_f16(
          false, A, false, B1, (short)0, accB[nt], false, false);
    }
  }
}
// preprocess fc1: A gathered from global f32, single padded k-chunk
template<int NT>
__device__ inline void mma_g16(v8f* acc, const float* __restrict__ s,
                               const _Float16* __restrict__ w) {
  const v16h A = ldA_g16(s);
#pragma unroll
  for (int nt = 0; nt < NT; ++nt) {
    const v16h B = ldB(w + (size_t)nt * 512);
    acc[nt] = __builtin_amdgcn_wmma_f32_16x16x32_f16(
        false, A, false, B, (short)0, acc[nt], false, false);
  }
}

// ---------- epilogues: D frag (m = r + 8*(lane>=16), n = lane&15) ----------
__device__ inline int f16_slot(int nG) {
  const int c = nG >> 5, kk = nG & 31;
  const int khi = (kk >> 4) & 1, rem16 = kk & 15;
  const int hi2 = (rem16 >> 3) & 1, rem8 = rem16 & 7;
  return c * 512 + hi2 * 256 + khi * 8 + rem8;
}

// ACT: 0 none, 1 relu, 2 tanh -> store f16 fragment-order
template<int NT, int ACT>
__device__ inline void ep_f16(const v8f* acc, _Float16* buf, int nbase) {
  const int l = laneid(), n = l & 15, mo = (l >= 16) ? 8 : 0;
#pragma unroll
  for (int nt = 0; nt < NT; ++nt) {
    const int sb = f16_slot(nbase + nt * 16 + n);
#pragma unroll
    for (int r = 0; r < 8; ++r) {
      float x = acc[nt][r];
      if (ACT == 1) x = fmaxf(x, 0.0f);
      if (ACT == 2) x = fast_tanh(x);
      buf[sb + (mo + r) * 16] = (_Float16)x;
    }
  }
}

// DBlock gate: t = tanh(fc1) * sigmoid(fc2)
template<int NT>
__device__ inline void ep_gate(const v8f* ta, const v8f* sg, _Float16* buf, int nbase) {
  const int l = laneid(), n = l & 15, mo = (l >= 16) ? 8 : 0;
#pragma unroll
  for (int nt = 0; nt < NT; ++nt) {
    const int sb = f16_slot(nbase + nt * 16 + n);
#pragma unroll
    for (int r = 0; r < 8; ++r) {
      const float x = fast_tanh(ta[nt][r]) * fast_sigmoid(sg[nt][r]);
      buf[sb + (mo + r) * 16] = (_Float16)x;
    }
  }
}

// RNN epilogue: tanh -> f16 state buffer + f32 d_seq output tile
template<int NT>
__device__ inline void ep_tanh_dual(const v8f* acc, _Float16* buf, float* __restrict__ g) {
  const int l = laneid(), n = l & 15, mo = (l >= 16) ? 8 : 0;
#pragma unroll
  for (int nt = 0; nt < NT; ++nt) {
    const int sb = f16_slot(nt * 16 + n);
#pragma unroll
    for (int r = 0; r < 8; ++r) {
      const float x = fast_tanh(acc[nt][r]);
      buf[sb + (mo + r) * 16] = (_Float16)x;
      g[(size_t)(mo + r) * 64 + nt * 16 + n] = x;
    }
  }
}

// Per-wave LDS (halves): U,X,TMP,A64,H,Z = 1024 each; T,T2 = 2048 each -> 10240 halves = 20 KB
#define WAVE_LDS_HALVES 10240

__global__ __launch_bounds__(128) void srnn_scan_kernel(Params p) {
  extern __shared__ _Float16 lds[];
  const int wave = threadIdx.x >> 5;
  const int lane = laneid();
  _Float16* base = lds + wave * WAVE_LDS_HALVES;
  _Float16* U   = base;
  _Float16* X   = base + 1024;
  _Float16* TMP = base + 2048;
  _Float16* A64 = base + 3072;
  _Float16* H   = base + 4096;
  _Float16* Z   = base + 5120;
  _Float16* T   = base + 6144;   // posterior t
  _Float16* T2  = base + 8192;   // prior t (separate buffer -> short mu/ls live ranges)

  const int tile = blockIdx.x * 4 + wave;   // 0..31
  const int row0 = tile * 16;
  const _Float16* __restrict__ W = p.ws;

  for (int i = lane; i < 512; i += 32) {
    ((unsigned int*)H)[i] = 0u;
    ((unsigned int*)Z)[i] = 0u;
  }
  for (int i = lane; i < SW_TOTAL / 1024; i += 32)
    __builtin_prefetch(W + (size_t)i * 1024, 0, 3);

  const int n  = lane & 15;
  const int mo = (lane >= 16) ? 8 : 0;

  // ---- hoist all bias values into per-lane registers (loop-invariant) ----
  float b_pu1[4], b_pu2[4], b_px1[4], b_px2[4], b_rnn[4], b_a1[4], b_a2[4];
  float b_pofc1[8], b_pofc2[8], b_pomu[4], b_pols[4];
  float b_prfc1[8], b_prfc2[8], b_prmu[4];
#pragma unroll
  for (int i = 0; i < 4; ++i) {
    b_pu1[i] = p.pu_b1[i * 16 + n];  b_pu2[i] = p.pu_b2[i * 16 + n];
    b_px1[i] = p.px_b1[i * 16 + n];  b_px2[i] = p.px_b2[i * 16 + n];
    b_rnn[i] = p.rnn_bih[i * 16 + n] + p.rnn_bhh[i * 16 + n];
    b_a1[i]  = p.a_b1[i * 16 + n];   b_a2[i]  = p.a_b2[i * 16 + n];
    b_pomu[i] = p.po_mu_b[i * 16 + n]; b_pols[i] = p.po_ls_b[i * 16 + n];
    b_prmu[i] = p.pr_mu_b[i * 16 + n];
  }
#pragma unroll
  for (int i = 0; i < 8; ++i) {
    b_pofc1[i] = p.po_fc1_b[i * 16 + n]; b_pofc2[i] = p.po_fc2_b[i * 16 + n];
    b_prfc1[i] = p.pr_fc1_b[i * 16 + n]; b_prfc2[i] = p.pr_fc2_b[i * 16 + n];
  }

  for (int t = 0; t < LL; ++t) {
    const float* __restrict__ extp   = p.ext   + ((size_t)t * BB + row0) * 16;
    const float* __restrict__ obsp   = p.obs   + ((size_t)t * BB + row0) * 16;
    const float* __restrict__ noisep = p.noise + ((size_t)t * BB + row0) * SS;
    const size_t obase = ((size_t)t * BB + row0) * SS;

    v8f a4[4];

    // ---- PreProcess(u) -> U ----
    acc_init_r<4>(a4, b_pu1);
    mma_g16<4>(a4, extp, W + OFF_PU_W1);
    ep_f16<4, 1>(a4, TMP, 0);
    acc_init_r<4>(a4, b_pu2);
    mma_loop<4, 2>(a4, src64(TMP), W + OFF_PU_W2);
    ep_f16<4, 1>(a4, U, 0);

    // ---- PreProcess(x) -> X ----
    acc_init_r<4>(a4, b_px1);
    mma_g16<4>(a4, obsp, W + OFF_PX_W1);
    ep_f16<4, 1>(a4, TMP, 0);
    acc_init_r<4>(a4, b_px2);
    mma_loop<4, 2>(a4, src64(TMP), W + OFF_PX_W2);
    ep_f16<4, 1>(a4, X, 0);

    // ---- RNN: h = tanh(U@Wih^T + H@Whh^T + b) -> H, d_seq ----
    acc_init_r<4>(a4, b_rnn);
    mma_loop<4, 2>(a4, src64(U), W + OFF_WIH);
    mma_loop<4, 2>(a4, src64(H), W + OFF_WHH);   // reads old H; stores after
    ep_tanh_dual<4>(a4, H, p.out_d + obase);

    // ---- a-layer: a = W2 @ tanh(W1 @ [d|x]) -> A64 ----
    acc_init_r<4>(a4, b_a1);
    mma_loop<4, 4>(a4, cat64(H, X), W + OFF_A_W1);
    ep_f16<4, 2>(a4, TMP, 0);
    acc_init_r<4>(a4, b_a2);
    mma_loop<4, 2>(a4, src64(TMP), W + OFF_A_W2);
    ep_f16<4, 0>(a4, A64, 0);

    // ---- posterior DBlock on [z|a]: two NT=4 halves -> T ----
    {
      const ASrc za = cat64(Z, A64);
      v8f f1[4], f2[4];
      acc_init_r<4>(f1, b_pofc1);
      acc_init_r<4>(f2, b_pofc2);
      mma_loop2<4, 4>(f1, f2, za, W + OFF_PO_FC1, W + OFF_PO_FC2);
      ep_gate<4>(f1, f2, T, 0);
      acc_init_r<4>(f1, b_pofc1 + 4);
      acc_init_r<4>(f2, b_pofc2 + 4);
      mma_loop2<4, 4>(f1, f2, za, W + OFF_PO_FC1 + HALF8, W + OFF_PO_FC2 + HALF8);
      ep_gate<4>(f1, f2, T, 64);
    }

    // ---- prior DBlock on [z|d], two NT=4 halves -> T2 ----
    {
      const ASrc zd = cat64(Z, H);
      v8f f1[4], f2[4];
      acc_init_r<4>(f1, b_prfc1);
      acc_init_r<4>(f2, b_prfc2);
      mma_loop2<4, 4>(f1, f2, zd, W + OFF_PR_FC1, W + OFF_PR_FC2);
      ep_gate<4>(f1, f2, T2, 0);
      acc_init_r<4>(f1, b_prfc1 + 4);
      acc_init_r<4>(f2, b_prfc2 + 4);
      mma_loop2<4, 4>(f1, f2, zd, W + OFF_PR_FC1 + HALF8, W + OFF_PR_FC2 + HALF8);
      ep_gate<4>(f1, f2, T2, 64);
    }

    // ---- heads (short live ranges now): res_mu/ls from T, prior mu from T2 ----
    v8f mu_r[4], ls_r[4];
    acc_init_r<4>(mu_r, b_pomu);
    acc_init_r<4>(ls_r, b_pols);
    mma_loop2<4, 4>(mu_r, ls_r, src128(T), W + OFF_PO_MU, W + OFF_PO_LS);

    v8f pmu[4];
    acc_init_r<4>(pmu, b_prmu);
    mma_loop<4, 4>(pmu, src128(T2), W + OFF_PR_MU);

    // ---- combine: mu, ls, z = mu + exp(0.5 ls) * eps ----
#pragma unroll
    for (int nt = 0; nt < 4; ++nt) {
      const int sb = f16_slot(nt * 16 + n);
#pragma unroll
      for (int r = 0; r < 8; ++r) {
        const int m   = mo + r;
        const int col = nt * 16 + n;
        const float mu = mu_r[nt][r] + pmu[nt][r];
        const float ls = ls_r[nt][r];
        const float ep = noisep[(size_t)m * SS + col];
        const float z  = mu + __expf(0.5f * ls) * ep;
        const size_t gi = obase + (size_t)m * SS + col;
        p.out_mu[gi] = mu;
        p.out_ls[gi] = ls;
        p.out_z[gi]  = z;
        Z[sb + m * 16] = (_Float16)z;   // state for t+1 (DS in-order per wave)
      }
    }
  }
}

// ---------------- one-shot weight swizzle: f32 [NO x K] -> f16 B-fragment order ----
struct SwDesc { const float* src; int dstOff; int NT; int KT; int kvalid; int ldk; };
struct SwParams { SwDesc d[15]; _Float16* dst; };

__global__ __launch_bounds__(256) void swizzle_kernel(SwParams sp) {
  const SwDesc d = sp.d[blockIdx.x];
  const int total = d.NT * d.KT * 512;
  for (int i = threadIdx.x; i < total; i += 256) {
    const int j    = i & 15;
    const int lane = (i >> 4) & 31;
    const int tidx = i >> 9;           // (nt*KT + kc)
    const int kc   = tidx % d.KT;
    const int nt   = tidx / d.KT;
    const int nn   = lane & 15;
    const int hi   = lane >> 4;
    const int k    = kc * 32 + (j >> 3) * 16 + hi * 8 + (j & 7);
    float v = 0.0f;
    if (k < d.kvalid) v = d.src[(size_t)(nt * 16 + nn) * d.ldk + k];
    sp.dst[d.dstOff + i] = (_Float16)v;
  }
}

extern "C" void kernel_launch(void* const* d_in, const int* in_sizes, int n_in,
                              void* d_out, int out_size, void* d_ws, size_t ws_size,
                              hipStream_t stream) {
  (void)in_sizes; (void)n_in; (void)out_size; (void)ws_size;
  const float* f[35];
  for (int i = 0; i < 35; ++i) f[i] = (const float*)d_in[i];

  _Float16* wsw = (_Float16*)d_ws;

  SwParams sp;
  sp.dst = wsw;
  sp.d[0]  = { f[3],  OFF_PU_W1,  4, 1, 16,  16  };
  sp.d[1]  = { f[5],  OFF_PU_W2,  4, 2, 64,  64  };
  sp.d[2]  = { f[7],  OFF_PX_W1,  4, 1, 16,  16  };
  sp.d[3]  = { f[9],  OFF_PX_W2,  4, 2, 64,  64  };
  sp.d[4]  = { f[11], OFF_WIH,    4, 2, 64,  64  };
  sp.d[5]  = { f[12], OFF_WHH,    4, 2, 64,  64  };
  sp.d[6]  = { f[15], OFF_A_W1,   4, 4, 128, 128 };
  sp.d[7]  = { f[17], OFF_A_W2,   4, 2, 64,  64  };
  sp.d[8]  = { f[19], OFF_PO_FC1, 8, 4, 128, 128 };
  sp.d[9]  = { f[21], OFF_PO_FC2, 8, 4, 128, 128 };
  sp.d[10] = { f[23], OFF_PO_MU,  4, 4, 128, 128 };
  sp.d[11] = { f[25], OFF_PO_LS,  4, 4, 128, 128 };
  sp.d[12] = { f[27], OFF_PR_FC1, 8, 4, 128, 128 };
  sp.d[13] = { f[29], OFF_PR_FC2, 8, 4, 128, 128 };
  sp.d[14] = { f[31], OFF_PR_MU,  4, 4, 128, 128 };
  hipLaunchKernelGGL(swizzle_kernel, dim3(15), dim3(256), 0, stream, sp);

  Params p;
  p.ext = f[0]; p.obs = f[1]; p.noise = f[2];
  p.pu_b1 = f[4];  p.pu_b2 = f[6];
  p.px_b1 = f[8];  p.px_b2 = f[10];
  p.rnn_bih = f[13]; p.rnn_bhh = f[14];
  p.a_b1 = f[16]; p.a_b2 = f[18];
  p.po_fc1_b = f[20]; p.po_fc2_b = f[22];
  p.po_mu_b  = f[24]; p.po_ls_b  = f[26];
  p.pr_fc1_b = f[28]; p.pr_fc2_b = f[30];
  p.pr_mu_b  = f[32];
  p.ws = wsw;

  float* out = (float*)d_out;
  const size_t N = (size_t)LL * BB * SS;
  p.out_mu = out;
  p.out_ls = out + N;
  p.out_z  = out + 2 * N;
  p.out_d  = out + 3 * N;

  const dim3 grid(8);
  const dim3 block(128);
  const size_t shmem = 4 * WAVE_LDS_HALVES * sizeof(_Float16);  // 80 KB
  hipLaunchKernelGGL(srnn_scan_kernel, grid, block, shmem, stream, p);
}
